// RNN_GRU_49400713839291
// MI455X (gfx1250) — matrix operand order repaired and checked by reading, not compiled, not run on hardware
//
#include <hip/hip_runtime.h>

// ---------------------------------------------------------------------------
// GRU (3 layers, zoneout) + dense head for MI455X (gfx1250, wave32, WMMA).
// B=16, T=4096, D=U=256.
//   * parallel WMMA GEMMs for input gates & dense head
//   * persistent 1-workgroup scan with recurrent weights resident in VGPRs,
//     ping-pong LDS h-buffer (1 barrier/step), gate loads overlapped w/ WMMA
// ---------------------------------------------------------------------------

typedef __attribute__((ext_vector_type(16))) _Float16 v16h;
typedef __attribute__((ext_vector_type(8)))  _Float16 v8h;
typedef __attribute__((ext_vector_type(8)))  float    v8f;

union F16x16 { v16h v; v8h h[2]; };

__device__ __forceinline__ v8f wmma_f16(v16h a, v16h b, v8f c) {
    // D = A(16x32 f16) x B(32x16 f16) + C(16x16 f32)
    return __builtin_amdgcn_wmma_f32_16x16x32_f16(
        /*neg_a=*/false, a, /*neg_b=*/false, b,
        /*c_mod=*/(short)0, c, /*reuse_a=*/false, /*reuse_b=*/false);
}

__device__ __forceinline__ float fsigmoid(float x) {
    return 1.0f / (1.0f + __expf(-x));
}
__device__ __forceinline__ float ftanh_fast(float x) {
    float e = __expf(2.0f * x);
    return 1.0f - 2.0f / (e + 1.0f);   // monotone, no NaN at +-inf
}

// ---------------------------------------------------------------------------
// Prep: convert f32 [K][N] row-major -> f16 [N][K] (transposed, K inner).
// ---------------------------------------------------------------------------
__global__ void conv_transpose_f16(const float* __restrict__ src,
                                   _Float16* __restrict__ dst,
                                   int K, int N) {
    int idx = blockIdx.x * 256 + threadIdx.x;
    if (idx >= K * N) return;
    int k = idx / N;
    int n = idx - k * N;
    dst[(size_t)n * K + k] = (_Float16)src[idx];
}

// ---------------------------------------------------------------------------
// Generic WMMA GEMM:  C[M,N] = A[M,256](f32) * Bt[N][256](f16)^T + bias[N]
// block = 128 threads (4 waves); grid = (M/16, N/64); wave w -> 16 cols.
// ---------------------------------------------------------------------------
__global__ __launch_bounds__(128)
void wmma_gemm_kernel(const float* __restrict__ A,
                      const _Float16* __restrict__ Bt,
                      const float* __restrict__ bias,
                      float* __restrict__ C,
                      int M, int N) {
    __shared__ _Float16 atile[16 * 256];   // 8 KB

    const int tid  = threadIdx.x;
    const int wave = tid >> 5;
    const int lane = tid & 31;
    const int cl   = lane & 15;
    const int hi   = lane >> 4;
    const size_t mbase = (size_t)blockIdx.x * 16;
    const int n0 = blockIdx.y * 64 + wave * 16;
    (void)M;

    // cooperative A tile load (f32 -> f16), row-major [16][256]
    for (int i = tid; i < 16 * 256; i += 128) {
        atile[i] = (_Float16)A[(mbase + (i >> 8)) * 256 + (i & 255)];
    }
    __syncthreads();

    const int ncol = n0 + cl;
    const float bv = bias[ncol];
    v8f acc = { bv, bv, bv, bv, bv, bv, bv, bv };

    const _Float16* brow = Bt + (size_t)ncol * 256 + hi * 8;
    const int arow = cl * 256 + hi * 8;

#pragma unroll
    for (int k = 0; k < 8; ++k) {
        F16x16 a, b;
        a.h[0] = *(const v8h*)(atile + arow + k * 32);        // K k*32+{0..7 | 8..15}
        a.h[1] = *(const v8h*)(atile + arow + k * 32 + 16);   // K k*32+{16..23 | 24..31}
        b.h[0] = *(const v8h*)(brow + k * 32);                // K k*32+{0..7 | 8..15}
        b.h[1] = *(const v8h*)(brow + k * 32 + 16);           // K k*32+{16..23 | 24..31}
        acc = wmma_f16(a.v, b.v, acc);
    }

    float* crow = C + (mbase + 8 * hi) * (size_t)N + ncol;
#pragma unroll
    for (int r = 0; r < 8; ++r) crow[(size_t)r * N] = acc[r];
}

// ---------------------------------------------------------------------------
// Persistent GRU scan: single workgroup, 512 threads = 16 waves.
// Wave w owns units [w*16, w*16+16): 3 gate column-tiles x 8 K-chunks of U
// in VGPRs (192 VGPRs of weights/lane).  h carried in ping-pong LDS (f16):
// step t reads hbuf[p], writes hbuf[p^1] -> one barrier per step.
// Gate accumulators are seeded with b_rec only; xz/xr are added AFTER the
// WMMA chain so the 24 global gate loads overlap the matrix work.
//   XG  : [16][T][768] f32 precomputed input gates (x@W + b_in)
//   Ut  : [768][256] f16 recurrent weights transposed
//   Y   : [16][T][256] f32 output (h_cand [+ residual])
// ---------------------------------------------------------------------------
#define ZONEOUT 0.1f

__global__ __launch_bounds__(512, 1)
void gru_scan_kernel(const float* __restrict__ XG,
                     const _Float16* __restrict__ Ut,
                     const float* __restrict__ brec,
                     const float* __restrict__ h0,
                     const float* __restrict__ Yprev,
                     float* __restrict__ Y,
                     int T, int addResidual) {
    __shared__ _Float16 hbuf[2][16 * 256];   // [pingpong][batch][unit], 16 KB

    const int tid  = threadIdx.x;
    const int wave = tid >> 5;
    const int lane = tid & 31;
    const int cl   = lane & 15;
    const int hi   = lane >> 4;
    const int u0   = wave * 16;
    const int ucol = u0 + cl;

    // --- recurrent weight fragments resident in VGPRs: 3 gates x 8 K-chunks
    v16h Bf[3][8];
#pragma unroll
    for (int g = 0; g < 3; ++g) {
        const _Float16* brow = Ut + (size_t)(g * 256 + ucol) * 256 + hi * 8;
#pragma unroll
        for (int k = 0; k < 8; ++k) {
            F16x16 b;
            b.h[0] = *(const v8h*)(brow + k * 32);
            b.h[1] = *(const v8h*)(brow + k * 32 + 16);
            Bf[g][k] = b.v;
        }
    }

    const float br0 = brec[0 * 256 + ucol];
    const float br1 = brec[1 * 256 + ucol];
    const float br2 = brec[2 * 256 + ucol];

    // zoneout-carried state for this lane's (batch r+8*hi, unit ucol) cells
    float hprev[8];
    const float h0v = h0[ucol];
#pragma unroll
    for (int r = 0; r < 8; ++r) hprev[r] = h0v;

    // initialize shared h buffer (broadcast h0 over batch)
    for (int i = tid; i < 16 * 256; i += 512) hbuf[0][i] = (_Float16)h0[i & 255];
    __syncthreads();

    const size_t bs_xg = (size_t)T * 768;
    const size_t bs_y  = (size_t)T * 256;
    const float* xgb = XG + (size_t)(8 * hi) * bs_xg + ucol;
    const float* ypb = Yprev + (size_t)(8 * hi) * bs_y + ucol;
    float*       yb  = Y + (size_t)(8 * hi) * bs_y + ucol;
    const int arow = cl * 256;

    int p = 0;
    for (int t = 0; t < T; ++t) {
        const size_t toff  = (size_t)t * 768;
        const size_t toffy = (size_t)t * 256;

        // issue this step's gate loads; they complete while WMMAs run
        float xz[8], xr[8], xh[8];
#pragma unroll
        for (int r = 0; r < 8; ++r) {
            const float* xp = xgb + (size_t)r * bs_xg + toff;
            xz[r] = xp[0];
            xr[r] = xp[256];
            xh[r] = xp[512];
            // prefetch next timestep's gate rows (global_prefetch_b8)
            __builtin_prefetch(xp + 768, 0, 1);
            __builtin_prefetch(xp + 768 + 256, 0, 1);
            __builtin_prefetch(xp + 768 + 512, 0, 1);
        }

        // rg = h @ U + b_rec  (24 WMMAs: 8 K-chunks x 3 independent gates)
        v8f accz = { br0, br0, br0, br0, br0, br0, br0, br0 };
        v8f accr = { br1, br1, br1, br1, br1, br1, br1, br1 };
        v8f acch = { br2, br2, br2, br2, br2, br2, br2, br2 };
        const _Float16* hb = hbuf[p];
#pragma unroll
        for (int k = 0; k < 8; ++k) {
            F16x16 a;
            const int ab = arow + k * 32 + hi * 8;
            a.h[0] = *(const v8h*)(hb + ab);
            a.h[1] = *(const v8h*)(hb + ab + 16);
            accz = wmma_f16(a.v, Bf[0][k], accz);
            accr = wmma_f16(a.v, Bf[1][k], accr);
            acch = wmma_f16(a.v, Bf[2][k], acch);
        }

        // elementwise GRU + zoneout; store h_cand (+ residual); write h(t+1)
        _Float16* hn_out = hbuf[p ^ 1];
#pragma unroll
        for (int r = 0; r < 8; ++r) {
            float z  = fsigmoid(xz[r] + accz[r]);
            float rg = fsigmoid(xr[r] + accr[r]);
            float hh = ftanh_fast(xh[r] + rg * acch[r]);
            float hc = z * hprev[r] + (1.0f - z) * hh;          // cell output
            float yv = hc;
            if (addResidual) yv += ypb[(size_t)r * bs_y + toffy];
            yb[(size_t)r * bs_y + toffy] = yv;
            float hn = hc + ZONEOUT * (hprev[r] - hc);          // zoneout carry
            hprev[r] = hn;
            hn_out[(r + 8 * hi) * 256 + ucol] = (_Float16)hn;
        }
        __syncthreads();   // h(t+1) buffer complete before next step reads it
        p ^= 1;
    }
}

// ---------------------------------------------------------------------------
// Host-side launch sequence
// ---------------------------------------------------------------------------
extern "C" void kernel_launch(void* const* d_in, const int* in_sizes, int n_in,
                              void* d_out, int out_size, void* d_ws, size_t ws_size,
                              hipStream_t stream) {
    (void)in_sizes; (void)n_in; (void)out_size; (void)ws_size;

    const float* x    = (const float*)d_in[0];
    const float* W[3]  = {(const float*)d_in[1], (const float*)d_in[5], (const float*)d_in[9]};
    const float* Uw[3] = {(const float*)d_in[2], (const float*)d_in[6], (const float*)d_in[10]};
    const float* bb[3] = {(const float*)d_in[3], (const float*)d_in[7], (const float*)d_in[11]};
    const float* h0[3] = {(const float*)d_in[4], (const float*)d_in[8], (const float*)d_in[12]};
    const float* Wd = (const float*)d_in[13];
    const float* bd = (const float*)d_in[14];

    const int B = 16, T = 4096;
    const int M = B * T;                       // 65536 rows

    // workspace carve-out
    char* ws = (char*)d_ws;
    size_t off = 0;
    auto carve = [&](size_t bytes) -> void* {
        void* p = ws + off;
        off += (bytes + 255) & ~(size_t)255;
        return p;
    };
    _Float16* Wt[3]; _Float16* Ut[3];
    for (int l = 0; l < 3; ++l) {
        Wt[l] = (_Float16*)carve((size_t)768 * 256 * 2);
        Ut[l] = (_Float16*)carve((size_t)768 * 256 * 2);
    }
    _Float16* Wdt = (_Float16*)carve((size_t)256 * 256 * 2);
    float* XG = (float*)carve((size_t)M * 768 * 4);   // input gates
    float* Ya = (float*)carve((size_t)M * 256 * 4);   // layer outputs ping
    float* Yb = (float*)carve((size_t)M * 256 * 4);   // layer outputs pong

    // --- prep: transpose+convert all weights to f16 [N][K]
    {
        const int n768 = 256 * 768;
        dim3 g((n768 + 255) / 256);
        for (int l = 0; l < 3; ++l) {
            conv_transpose_f16<<<g, 256, 0, stream>>>(W[l],  Wt[l], 256, 768);
            conv_transpose_f16<<<g, 256, 0, stream>>>(Uw[l], Ut[l], 256, 768);
        }
        const int n256 = 256 * 256;
        conv_transpose_f16<<<dim3((n256 + 255) / 256), 256, 0, stream>>>(Wd, Wdt, 256, 256);
    }

    dim3 gateGrid(M / 16, 768 / 64);
    dim3 headGrid(M / 16, 256 / 64);

    // layer 0: xg = x@W0 + b_in ; scan -> Ya
    wmma_gemm_kernel<<<gateGrid, 128, 0, stream>>>(x, Wt[0], bb[0], XG, M, 768);
    gru_scan_kernel<<<1, 512, 0, stream>>>(XG, Ut[0], bb[0] + 768, h0[0], Ya, Ya, T, 0);

    // layer 1: xg = Ya@W1 + b_in ; scan (+residual Ya) -> Yb
    wmma_gemm_kernel<<<gateGrid, 128, 0, stream>>>(Ya, Wt[1], bb[1], XG, M, 768);
    gru_scan_kernel<<<1, 512, 0, stream>>>(XG, Ut[1], bb[1] + 768, h0[1], Ya, Yb, T, 1);

    // layer 2: xg = Yb@W2 + b_in ; scan (+residual Yb) -> Ya
    wmma_gemm_kernel<<<gateGrid, 128, 0, stream>>>(Yb, Wt[2], bb[2], XG, M, 768);
    gru_scan_kernel<<<1, 512, 0, stream>>>(XG, Ut[2], bb[2] + 768, h0[2], Yb, Ya, T, 1);

    // dense head: out = Ya@Wd + bd
    wmma_gemm_kernel<<<headGrid, 128, 0, stream>>>(Ya, Wdt, bd, (float*)d_out, M, 256);
}
